// Generator_86466281603773
// MI455X (gfx1250) — matrix operand order repaired and checked by reading, not compiled, not run on hardware
//
#include <hip/hip_runtime.h>
#include <hip/hip_bf16.h>

// ---------------------------------------------------------------------------
// Show-Attend-Tell LSTM decoder on MI455X (gfx1250), wave32.
// GEMMs: V_WMMA_F32_16X16X32_F16, one wave = 16x64 tile (4 WMMAs / k-step),
// software-pipelined k-loop. Alpha row staged to LDS via the Tensor Data
// Mover (tensor_load_to_lds + s_wait_tensorcnt) where available.
// ---------------------------------------------------------------------------

#define B_   64
#define P_   196
#define ENC_ 2048
#define E_   512
#define A_   512
#define U_   512
#define V_   10000
#define S_   22
#define T_   21
#define XH_  (E_ + ENC_ + U_)   // 3072 : [emb | awe*beta | h]

typedef __attribute__((ext_vector_type(16))) _Float16 v16h;
typedef __attribute__((ext_vector_type(8)))  _Float16 v8h;
typedef __attribute__((ext_vector_type(8)))  float    v8f;
typedef __attribute__((ext_vector_type(4)))  unsigned int v4u;
typedef __attribute__((ext_vector_type(8)))  int      v8i;
typedef __attribute__((ext_vector_type(4)))  int      v4i;

#if defined(__HIP_DEVICE_COMPILE__)
#if __has_builtin(__builtin_amdgcn_tensor_load_to_lds) && \
    __has_builtin(__builtin_amdgcn_s_wait_tensorcnt)
#define HAVE_TDM 1
#endif
#endif

// ---- fragment loaders (CDNA5 ISA 7.12.2 wave32 layouts) -------------------
// A frag: lane holds K = {0..7}+16*{0,1} shifted by hsel*8 (ptr pre-offset)
__device__ __forceinline__ v16h ld_afrag(const _Float16* p) {
  v8h lo = *(const v8h*)p;
  v8h hi = *(const v8h*)(p + 16);
  v16h r;
#pragma unroll
  for (int i = 0; i < 8; ++i) { r[i] = lo[i]; r[i + 8] = hi[i]; }
  return r;
}
// B frag: lane holds K = {0..7}+16*{0,1} shifted by hsel*8 (ptr pre-offset)
__device__ __forceinline__ v16h ld_bfrag(const _Float16* p) {
  v8h lo = *(const v8h*)p;
  v8h hi = *(const v8h*)(p + 16);
  v16h r;
#pragma unroll
  for (int i = 0; i < 8; ++i) { r[i] = lo[i]; r[i + 8] = hi[i]; }
  return r;
}
// A frag from fp32 source (converted on the fly)
__device__ __forceinline__ v16h ld_afrag_f32(const float* p) {
  v8f lo = *(const v8f*)p;
  v8f hi = *(const v8f*)(p + 16);
  v16h r;
#pragma unroll
  for (int i = 0; i < 8; ++i) {
    r[i] = (_Float16)lo[i];
    r[i + 8] = (_Float16)hi[i];
  }
  return r;
}

#define WMMA(acc, a, b) \
  acc = __builtin_amdgcn_wmma_f32_16x16x32_f16(false, (a), false, (b), (short)0, (acc), false, false)

// ---------------------------------------------------------------------------
// Generic WMMA GEMM: C[M,N] = A_f16[M,K] @ Bt_f16[N,K]^T + bias[N]
// One wave = 16 rows x 64 cols (4 n-tiles); double-buffered k-loop.
// M mult of 16, K mult of 32; N arbitrary mult of 16 (tail group clamps).
// ---------------------------------------------------------------------------
__global__ void wmma_gemm_f16(const _Float16* __restrict__ A,
                              const _Float16* __restrict__ Bt,
                              const float* __restrict__ bias,
                              float* __restrict__ C,
                              int M, int N, int K) {
  int ntn4 = (N + 63) >> 6;
  int wave = (blockIdx.x * blockDim.x + threadIdx.x) >> 5;
  if (wave >= (M >> 4) * ntn4) return;
  int m0  = (wave / ntn4) << 4;
  int ng0 = (wave % ntn4) << 6;
  int lane = threadIdx.x & 31;
  int hsel = lane >> 4, l16 = lane & 15;
  const _Float16* arow = A + (size_t)(m0 + l16) * K + hsel * 8;
  const _Float16* bcol[4];
#pragma unroll
  for (int j = 0; j < 4; ++j) {
    int nb = ng0 + j * 16;
    if (nb > N - 16) nb = N - 16;          // clamp (recompute, store skipped)
    bcol[j] = Bt + (size_t)(nb + l16) * K + hsel * 8;
  }
  v8f acc0 = {}, acc1 = {}, acc2 = {}, acc3 = {};
  v16h ac  = ld_afrag(arow);
  v16h bc0 = ld_bfrag(bcol[0]), bc1 = ld_bfrag(bcol[1]);
  v16h bc2 = ld_bfrag(bcol[2]), bc3 = ld_bfrag(bcol[3]);
  for (int k0 = 0; k0 < K; k0 += 32) {
    int kn = (k0 + 32 < K) ? (k0 + 32) : 0;   // branchless wrap on last iter
    v16h an  = ld_afrag(arow + kn);
    v16h bn0 = ld_bfrag(bcol[0] + kn), bn1 = ld_bfrag(bcol[1] + kn);
    v16h bn2 = ld_bfrag(bcol[2] + kn), bn3 = ld_bfrag(bcol[3] + kn);
    __builtin_prefetch(arow + kn + 32, 0, 1);
    WMMA(acc0, ac, bc0);
    WMMA(acc1, ac, bc1);
    WMMA(acc2, ac, bc2);
    WMMA(acc3, ac, bc3);
    ac = an; bc0 = bn0; bc1 = bn1; bc2 = bn2; bc3 = bn3;
  }
  v8f accs[4] = {acc0, acc1, acc2, acc3};
#pragma unroll
  for (int j = 0; j < 4; ++j) {
    int nb = ng0 + j * 16;
    if (nb + 16 > N) break;                  // skip clamped duplicate tiles
#pragma unroll
    for (int r = 0; r < 8; ++r) {
      int m = m0 + r + hsel * 8;             // C/D layout: VGPR r -> row r(+8)
      int n = nb + l16;
      C[(size_t)m * N + n] = accs[j][r] + bias[n];
    }
  }
}

// ---------------------------------------------------------------------------
// att1 = enc_sorted @ W_att_enc : fp32 A with per-row batch reorder.
// M=B*P=12544, N=512, K=2048. Same 16x64 pipelined scheme.
// ---------------------------------------------------------------------------
__global__ void wmma_gemm_att1(const float* __restrict__ enc,
                               const int* __restrict__ order,
                               const _Float16* __restrict__ Bt,
                               const float* __restrict__ bias,
                               float* __restrict__ C) {
  const int M = B_ * P_, N = A_, K = ENC_;
  int ntn4 = N >> 6;                         // 8
  int wave = (blockIdx.x * blockDim.x + threadIdx.x) >> 5;
  if (wave >= (M >> 4) * ntn4) return;
  int m0  = (wave / ntn4) << 4;
  int ng0 = (wave % ntn4) << 6;
  int lane = threadIdx.x & 31;
  int hsel = lane >> 4, l16 = lane & 15;
  int mr = m0 + l16;
  int b  = mr / P_, p = mr % P_;
  const float* arow = enc + ((size_t)order[b] * P_ + p) * (size_t)K + hsel * 8;
  const _Float16* bcol[4];
#pragma unroll
  for (int j = 0; j < 4; ++j)
    bcol[j] = Bt + (size_t)(ng0 + j * 16 + l16) * K + hsel * 8;
  v8f acc0 = {}, acc1 = {}, acc2 = {}, acc3 = {};
  v16h ac  = ld_afrag_f32(arow);
  v16h bc0 = ld_bfrag(bcol[0]), bc1 = ld_bfrag(bcol[1]);
  v16h bc2 = ld_bfrag(bcol[2]), bc3 = ld_bfrag(bcol[3]);
  for (int k0 = 0; k0 < K; k0 += 32) {
    int kn = (k0 + 32 < K) ? (k0 + 32) : 0;
    v16h an  = ld_afrag_f32(arow + kn);
    v16h bn0 = ld_bfrag(bcol[0] + kn), bn1 = ld_bfrag(bcol[1] + kn);
    v16h bn2 = ld_bfrag(bcol[2] + kn), bn3 = ld_bfrag(bcol[3] + kn);
    __builtin_prefetch(arow + kn + 32, 0, 1);
    WMMA(acc0, ac, bc0);
    WMMA(acc1, ac, bc1);
    WMMA(acc2, ac, bc2);
    WMMA(acc3, ac, bc3);
    ac = an; bc0 = bn0; bc1 = bn1; bc2 = bn2; bc3 = bn3;
  }
  v8f accs[4] = {acc0, acc1, acc2, acc3};
#pragma unroll
  for (int j = 0; j < 4; ++j) {
#pragma unroll
    for (int r = 0; r < 8; ++r) {
      int m = m0 + r + hsel * 8;
      int n = ng0 + j * 16 + l16;
      C[(size_t)m * N + n] = accs[j][r] + bias[n];
    }
  }
}

// ---------------------------------------------------------------------------
// Setup kernels
// ---------------------------------------------------------------------------
__global__ void sort_kernel(const int* __restrict__ lens_in,
                            const int* __restrict__ seqs_in,
                            int* __restrict__ order, int* __restrict__ ilen,
                            int* __restrict__ seqs_sorted,
                            float* __restrict__ out_seqs,
                            float* __restrict__ out_ilen,
                            float* __restrict__ out_order) {
  __shared__ int lens[B_];
  int i = threadIdx.x;
  lens[i] = lens_in[i];
  __syncthreads();
  int rank = 0, li = lens[i];
  for (int j = 0; j < B_; ++j) {
    int lj = lens[j];
    if (lj > li || (lj == li && j < i)) rank++;
  }
  order[rank] = i;
  __syncthreads();
  int o = order[i];
  int il = lens[o] - 1;
  ilen[i] = il;
  out_ilen[i]  = (float)il;
  out_order[i] = (float)o;
  for (int s = 0; s < S_; ++s) {
    int tok = seqs_in[o * S_ + s];
    seqs_sorted[i * S_ + s] = tok;
    out_seqs[i * S_ + s] = (float)tok;
  }
}

__global__ void mean_enc_kernel(const float* __restrict__ enc,
                                const int* __restrict__ order,
                                _Float16* __restrict__ mean_f16) {
  int tid = blockIdx.x * blockDim.x + threadIdx.x;
  if (tid >= B_ * ENC_) return;
  int b = tid / ENC_, e = tid % ENC_;
  const float* base = enc + ((size_t)order[b] * P_) * ENC_ + e;
  float s = 0.f;
  for (int p = 0; p < P_; ++p) s += base[(size_t)p * ENC_];
  mean_f16[tid] = (_Float16)(s * (1.0f / (float)P_));
}

__global__ void transpose_f16(const float* __restrict__ in,
                              _Float16* __restrict__ out, int K, int N) {
  int tid = blockIdx.x * blockDim.x + threadIdx.x;
  if (tid >= K * N) return;
  int k = tid / N, n = tid % N;
  out[(size_t)n * K + k] = (_Float16)in[tid];
}

__global__ void build_wxh(const float* __restrict__ Wx,
                          const float* __restrict__ Wh,
                          _Float16* __restrict__ out) {
  int tid = blockIdx.x * blockDim.x + threadIdx.x;
  if (tid >= (4 * U_) * XH_) return;
  int n = tid / XH_, k = tid % XH_;
  float v = (k < E_ + ENC_) ? Wx[(size_t)k * (4 * U_) + n]
                            : Wh[(size_t)(k - (E_ + ENC_)) * (4 * U_) + n];
  out[tid] = (_Float16)v;
}

__global__ void f32_to_f16(const float* __restrict__ in,
                           _Float16* __restrict__ out, int n) {
  int tid = blockIdx.x * blockDim.x + threadIdx.x;
  if (tid < n) out[tid] = (_Float16)in[tid];
}

// ---------------------------------------------------------------------------
// Per-step kernels
// ---------------------------------------------------------------------------
__global__ void att_e_kernel(const float* __restrict__ att1,
                             const float* __restrict__ att2,
                             const float* __restrict__ Wfull,
                             const float* __restrict__ bfull,
                             float* __restrict__ e) {
  int row  = (blockIdx.x * blockDim.x + threadIdx.x) >> 5;
  int lane = threadIdx.x & 31;
  if (row >= B_ * P_) return;
  int b = row / P_;
  const float* a1 = att1 + (size_t)row * A_;
  const float* a2 = att2 + (size_t)b * A_;
  float s = 0.f;
  for (int i = lane; i < A_; i += 32) {
    float v = a1[i] + a2[i];
    v = v > 0.f ? v : 0.f;
    s += v * Wfull[i];
  }
  for (int off = 16; off; off >>= 1) s += __shfl_xor(s, off, 32);
  if (lane == 0) e[row] = s + bfull[0];
}

__global__ void alpha_softmax(const float* __restrict__ e,
                              const int* __restrict__ ilen, int t,
                              float* __restrict__ alpha,
                              float* __restrict__ out_alpha) {
  int b = blockIdx.x, tid = threadIdx.x;
  const float* row = e + b * P_;
  __shared__ float red[256];
  float m = -1e30f;
  for (int p = tid; p < P_; p += 256) m = fmaxf(m, row[p]);
  red[tid] = m; __syncthreads();
  for (int s = 128; s; s >>= 1) { if (tid < s) red[tid] = fmaxf(red[tid], red[tid + s]); __syncthreads(); }
  m = red[0]; __syncthreads();
  float ss = 0.f;
  for (int p = tid; p < P_; p += 256) ss += __expf(row[p] - m);
  red[tid] = ss; __syncthreads();
  for (int s = 128; s; s >>= 1) { if (tid < s) red[tid] += red[tid + s]; __syncthreads(); }
  float inv = 1.0f / red[0];
  bool act = t < ilen[b];
  for (int p = tid; p < P_; p += 256) {
    float a = __expf(row[p] - m) * inv;
    alpha[b * P_ + p] = a;
    out_alpha[((size_t)b * T_ + t) * P_ + p] = act ? a : 0.0f;
  }
}

// xh_f16[b, 0:512]=emb ; [512:2560]=awe*sigmoid(beta) ; [2560:3072]=h
// Alpha row staged into LDS via the Tensor Data Mover where available.
__global__ void build_xh(const float* __restrict__ emb_table,
                         const int* __restrict__ seqs_sorted, int t,
                         const float* __restrict__ enc,
                         const int* __restrict__ order,
                         const float* __restrict__ alpha,
                         const float* __restrict__ beta_raw,
                         const float* __restrict__ h,
                         _Float16* __restrict__ xh) {
  __shared__ float sh_alpha[256];            // LDS offset 0 (only shared obj)
  int tid = blockIdx.x * blockDim.x + threadIdx.x;
  int b = (int)(blockIdx.x / (XH_ / 256));   // XH_%256==0 -> one b per block
#ifdef HAVE_TDM
  if (threadIdx.x < 32) {                    // wave 0 issues the DMA
    unsigned long long ga =
        (unsigned long long)(const void*)(alpha + (size_t)b * P_);
    v4u g0 = {0u, 0u, 0u, 0u};
    g0[0] = 1u;                              // count = 1 descriptor
    g0[1] = 0u;                              // lds_addr = 0 (sh_alpha)
    g0[2] = (unsigned)ga;                    // global_addr[31:0]
    g0[3] = (unsigned)((ga >> 32) & 0x01FFFFFFull) | 0x80000000u; // type=2
    v8i g1 = {0, 0, 0, 0, 0, 0, 0, 0};
    g1[0] = (int)(2u << 16);                 // data_size = 4 bytes
    g1[1] = (int)((unsigned)P_ << 16);       // tensor_dim0[15:0]
    g1[2] = (int)(1u << 16);                 // tensor_dim1 = 1
    g1[3] = (int)((unsigned)P_ << 16);       // tile_dim0 = 196
    g1[4] = 1;                               // tile_dim1 = 1
    g1[5] = P_;                              // tensor_dim0_stride
    v4i gz = {0, 0, 0, 0};
#if defined(__clang_major__) && (__clang_major__ >= 23)
    v8i gz8 = {0, 0, 0, 0, 0, 0, 0, 0};
    __builtin_amdgcn_tensor_load_to_lds(g0, g1, gz, gz, gz8, 0);
#else
    __builtin_amdgcn_tensor_load_to_lds(g0, g1, gz, gz, 0);
#endif
    __builtin_amdgcn_s_wait_tensorcnt(0);
  }
  __syncthreads();
#else
  if (threadIdx.x < P_) sh_alpha[threadIdx.x] = alpha[b * P_ + threadIdx.x];
  __syncthreads();
#endif
  if (tid >= B_ * XH_) return;
  int j = tid % XH_;
  float val;
  if (j < E_) {
    int tok = seqs_sorted[b * S_ + t];
    val = emb_table[(size_t)tok * E_ + j];
  } else if (j < E_ + ENC_) {
    int e = j - E_;
    const float* ebase = enc + ((size_t)order[b] * P_) * ENC_ + e;
    float s = 0.f;
    for (int p = 0; p < P_; ++p) s += ebase[(size_t)p * ENC_] * sh_alpha[p];
    float br  = beta_raw[b * ENC_ + e];
    float sig = 1.0f / (1.0f + __expf(-br));
    val = s * sig;
  } else {
    val = h[b * U_ + (j - E_ - ENC_)];
  }
  xh[tid] = (_Float16)val;
}

__global__ void lstm_step(const float* __restrict__ z,
                          const int* __restrict__ ilen, int t,
                          float* __restrict__ h, float* __restrict__ c,
                          _Float16* __restrict__ c_f16) {
  int tid = blockIdx.x * blockDim.x + threadIdx.x;
  if (tid >= B_ * U_) return;
  int b = tid / U_, u = tid % U_;
  const float* zb = z + (size_t)b * (4 * U_);
  float zi = zb[u], zf = zb[U_ + u], zg = zb[2 * U_ + u], zo = zb[3 * U_ + u];
  float si = 1.f / (1.f + __expf(-zi));
  float sf = 1.f / (1.f + __expf(-zf));
  float so = 1.f / (1.f + __expf(-zo));
  float g  = tanhf(zg);
  float cold = c[tid], hold = h[tid];
  float cn = sf * cold + si * g;
  float hn = so * tanhf(cn);
  bool act = t < ilen[b];
  float c2 = act ? cn : cold;
  float h2 = act ? hn : hold;
  c[tid] = c2; h[tid] = h2;
  c_f16[tid] = (_Float16)c2;
}

__global__ void pred_softmax(const float* __restrict__ logits,
                             const int* __restrict__ ilen, int t,
                             float* __restrict__ out_pred) {
  int b = blockIdx.x, tid = threadIdx.x;
  const float* row = logits + (size_t)b * V_;
  __shared__ float red[256];
  float m = -1e30f;
  for (int v = tid; v < V_; v += 256) m = fmaxf(m, row[v]);
  red[tid] = m; __syncthreads();
  for (int s = 128; s; s >>= 1) { if (tid < s) red[tid] = fmaxf(red[tid], red[tid + s]); __syncthreads(); }
  m = red[0]; __syncthreads();
  float ss = 0.f;
  for (int v = tid; v < V_; v += 256) ss += __expf(row[v] - m);
  red[tid] = ss; __syncthreads();
  for (int s = 128; s; s >>= 1) { if (tid < s) red[tid] += red[tid + s]; __syncthreads(); }
  float inv = 1.0f / red[0];
  bool act = t < ilen[b];
  float* dst = out_pred + ((size_t)b * T_ + t) * V_;
  for (int v = tid; v < V_; v += 256)
    dst[v] = act ? __expf(row[v] - m) * inv : 0.0f;
}

// ---------------------------------------------------------------------------
extern "C" void kernel_launch(void* const* d_in, const int* in_sizes, int n_in,
                              void* d_out, int out_size, void* d_ws, size_t ws_size,
                              hipStream_t stream) {
  const float* enc        = (const float*)d_in[0];
  const int*   sequences  = (const int*)  d_in[1];
  const int*   seq_lens   = (const int*)  d_in[2];
  const float* emb_table  = (const float*)d_in[3];
  const float* W_att_enc  = (const float*)d_in[4];
  const float* b_att_enc  = (const float*)d_in[5];
  const float* W_att_gen  = (const float*)d_in[6];
  const float* b_att_gen  = (const float*)d_in[7];
  const float* W_att_full = (const float*)d_in[8];
  const float* b_att_full = (const float*)d_in[9];
  const float* W_x        = (const float*)d_in[10];
  const float* W_h        = (const float*)d_in[11];
  const float* b_lstm     = (const float*)d_in[12];
  const float* W_init_m   = (const float*)d_in[13];
  const float* b_init_m   = (const float*)d_in[14];
  const float* W_init_c   = (const float*)d_in[15];
  const float* b_init_c   = (const float*)d_in[16];
  const float* W_beta     = (const float*)d_in[17];
  const float* b_beta     = (const float*)d_in[18];
  const float* W_out      = (const float*)d_in[19];
  const float* b_out      = (const float*)d_in[20];

  float* out_pred  = (float*)d_out;                       // [B,T,V]
  float* out_alpha = out_pred  + (size_t)B_ * T_ * V_;    // [B,T,P]
  float* out_seqs  = out_alpha + (size_t)B_ * T_ * P_;    // [B,S]
  float* out_ilen  = out_seqs  + (size_t)B_ * S_;         // [B]
  float* out_order = out_ilen  + B_;                      // [B]

  char* wsp = (char*)d_ws;
  auto carve = [&](size_t bytes) -> char* {
    char* p = wsp;
    wsp += (bytes + 255) & ~(size_t)255;
    return p;
  };
  float*    att1      = (float*)   carve((size_t)B_ * P_ * A_ * 4);
  _Float16* WattEnc_t = (_Float16*)carve((size_t)A_ * ENC_ * 2);
  _Float16* WattGen_t = (_Float16*)carve((size_t)A_ * U_ * 2);
  _Float16* Wbeta_t   = (_Float16*)carve((size_t)ENC_ * U_ * 2);
  _Float16* Winitm_t  = (_Float16*)carve((size_t)U_ * ENC_ * 2);
  _Float16* Winitc_t  = (_Float16*)carve((size_t)U_ * ENC_ * 2);
  _Float16* Wxh_t     = (_Float16*)carve((size_t)(4 * U_) * XH_ * 2);
  _Float16* Wout_t    = (_Float16*)carve((size_t)V_ * U_ * 2);
  _Float16* mean_f16  = (_Float16*)carve((size_t)B_ * ENC_ * 2);
  float*    h_st      = (float*)   carve((size_t)B_ * U_ * 4);
  float*    c_st      = (float*)   carve((size_t)B_ * U_ * 4);
  _Float16* c_f16     = (_Float16*)carve((size_t)B_ * U_ * 2);
  float*    att2      = (float*)   carve((size_t)B_ * A_ * 4);
  float*    e_buf     = (float*)   carve((size_t)B_ * P_ * 4);
  float*    alpha     = (float*)   carve((size_t)B_ * P_ * 4);
  float*    beta_raw  = (float*)   carve((size_t)B_ * ENC_ * 4);
  _Float16* xh        = (_Float16*)carve((size_t)B_ * XH_ * 2);
  float*    z_buf     = (float*)   carve((size_t)B_ * (4 * U_) * 4);
  float*    logits    = (float*)   carve((size_t)B_ * V_ * 4);
  int*      order_i   = (int*)     carve(B_ * 4);
  int*      ilen_i    = (int*)     carve(B_ * 4);
  int*      seqs_srt  = (int*)     carve((size_t)B_ * S_ * 4);
  (void)ws_size; (void)n_in; (void)in_sizes; (void)out_size;

  auto grid1d = [](size_t n) { return (unsigned)((n + 255) / 256); };
  // waves = (M/16) * ceil(N/64); 8 waves per 256-thread block
  auto gemm_grid = [](int M, int N) {
    return (unsigned)(((M / 16) * ((N + 63) / 64) + 7) / 8);
  };

  // ---- setup ----
  sort_kernel<<<1, B_, 0, stream>>>(seq_lens, sequences, order_i, ilen_i,
                                    seqs_srt, out_seqs, out_ilen, out_order);
  mean_enc_kernel<<<grid1d((size_t)B_ * ENC_), 256, 0, stream>>>(enc, order_i, mean_f16);

  transpose_f16<<<grid1d((size_t)ENC_ * A_), 256, 0, stream>>>(W_att_enc, WattEnc_t, ENC_, A_);
  transpose_f16<<<grid1d((size_t)U_ * A_),   256, 0, stream>>>(W_att_gen, WattGen_t, U_, A_);
  transpose_f16<<<grid1d((size_t)U_ * ENC_), 256, 0, stream>>>(W_beta,    Wbeta_t,   U_, ENC_);
  transpose_f16<<<grid1d((size_t)ENC_ * U_), 256, 0, stream>>>(W_init_m,  Winitm_t,  ENC_, U_);
  transpose_f16<<<grid1d((size_t)ENC_ * U_), 256, 0, stream>>>(W_init_c,  Winitc_t,  ENC_, U_);
  transpose_f16<<<grid1d((size_t)U_ * V_),   256, 0, stream>>>(W_out,     Wout_t,    U_, V_);
  build_wxh<<<grid1d((size_t)(4 * U_) * XH_), 256, 0, stream>>>(W_x, W_h, Wxh_t);

  wmma_gemm_att1<<<gemm_grid(B_ * P_, A_), 256, 0, stream>>>(enc, order_i, WattEnc_t,
                                                             b_att_enc, att1);
  wmma_gemm_f16<<<gemm_grid(B_, U_), 256, 0, stream>>>(mean_f16, Winitm_t, b_init_m,
                                                       h_st, B_, U_, ENC_);
  wmma_gemm_f16<<<gemm_grid(B_, U_), 256, 0, stream>>>(mean_f16, Winitc_t, b_init_c,
                                                       c_st, B_, U_, ENC_);
  f32_to_f16<<<grid1d((size_t)B_ * U_), 256, 0, stream>>>(c_st, c_f16, B_ * U_);

  // ---- 21 sequential decode steps ----
  for (int t = 0; t < T_; ++t) {
    wmma_gemm_f16<<<gemm_grid(B_, A_), 256, 0, stream>>>(c_f16, WattGen_t, b_att_gen,
                                                         att2, B_, A_, U_);
    att_e_kernel<<<(B_ * P_ * 32 + 255) / 256, 256, 0, stream>>>(att1, att2, W_att_full,
                                                                 b_att_full, e_buf);
    alpha_softmax<<<B_, 256, 0, stream>>>(e_buf, ilen_i, t, alpha, out_alpha);
    wmma_gemm_f16<<<gemm_grid(B_, ENC_), 256, 0, stream>>>(c_f16, Wbeta_t, b_beta,
                                                           beta_raw, B_, ENC_, U_);
    build_xh<<<grid1d((size_t)B_ * XH_), 256, 0, stream>>>(emb_table, seqs_srt, t, enc,
                                                           order_i, alpha, beta_raw,
                                                           h_st, xh);
    wmma_gemm_f16<<<gemm_grid(B_, 4 * U_), 256, 0, stream>>>(xh, Wxh_t, b_lstm,
                                                             z_buf, B_, 4 * U_, XH_);
    lstm_step<<<grid1d((size_t)B_ * U_), 256, 0, stream>>>(z_buf, ilen_i, t,
                                                           h_st, c_st, c_f16);
    wmma_gemm_f16<<<gemm_grid(B_, V_), 256, 0, stream>>>(c_f16, Wout_t, b_out,
                                                         logits, B_, V_, U_);
    pred_softmax<<<B_, 256, 0, stream>>>(logits, ilen_i, t, out_pred);
  }
}
